// TemporalFusion_43619687858291
// MI455X (gfx1250) — matrix operand order repaired and checked by reading, not compiled, not run on hardware
//
#include <hip/hip_runtime.h>

// ---------------------------------------------------------------- constants
constexpr int TT  = 10;
constexpr int CHN = 16;
constexpr int HH  = 16;
constexpr int WWp = 264;
constexpr int HWp = HH * WWp;          // 4224
constexpr int DD  = CHN * HH * WWp;    // 67584

constexpr size_t TB_F32_BYTES  = (size_t)TT * TT * DD * 4;   // 27,033,600
constexpr size_t TB_BF16_OFF   = TB_F32_BYTES;               // 27,033,600
constexpr size_t TB_BF16_BYTES = (size_t)TT * TT * DD * 2;   // 13,516,800
constexpr size_t SCORES_OFF    = TB_BF16_OFF + TB_BF16_BYTES;// 40,550,400
constexpr size_t WGT_OFF       = SCORES_OFF + (size_t)TT * 256 * 4;

typedef __attribute__((ext_vector_type(16))) __bf16 v16bf;
typedef __attribute__((ext_vector_type(8)))  __bf16 v8bf;
typedef __attribute__((ext_vector_type(8)))  float  v8f;

// ---------------------------------------------------------------- kernel 1:
// bilinear warp of ff_sp[ts] by pose = P[ts-1]-P[bs]; writes tb slot (bs, 9-ts)
// in both f32 and bf16. Invalid pairs (ts > bs) write zeros (slots 0..8 are
// each written exactly once across the 90 pairs; slot 9 is written by k2).
__global__ void tf_warp_kernel(const float* __restrict__ ff,
                               const float* __restrict__ dp,
                               float* __restrict__ tb,
                               __bf16* __restrict__ tbh) {
    const int p  = blockIdx.y;           // 0..89
    const int bs = p / (TT - 1);
    const int ts = p % (TT - 1) + 1;

    __shared__ float sh[4];              // dx, dy, cos, sin
    if (threadIdx.x == 0) {
        float px = 0.f, py = 0.f, pw = 0.f;   // P[bs]
        float qx = 0.f, qy = 0.f, qw = 0.f;   // P[ts-1]
        float cx = 0.f, cy = 0.f, cw = 0.f;   // running cumsum = P[i]
        for (int i = 0; i < TT; ++i) {
            if (i == bs)     { px = cx; py = cy; pw = cw; }
            if (i == ts - 1) { qx = cx; qy = cy; qw = cw; }
            cx += dp[i * 3 + 0];
            cy += dp[i * 3 + 1];
            cw += dp[i * 3 + 2];
        }
        const float yaw = qw - pw;            // pose = P[ts-1] - P[bs]
        sh[0] = qx - px;
        sh[1] = qy - py;
        sh[2] = __cosf(yaw);
        sh[3] = __sinf(yaw);
    }
    __syncthreads();

    const int idx = blockIdx.x * blockDim.x + threadIdx.x;   // 0..HWp-1
    if (idx >= HWp) return;
    const int y = idx / WWp;
    const int x = idx - y * WWp;

    const size_t slot = (size_t)bs * TT + (TT - 1 - ts);
    float*  out  = tb  + slot * DD;
    __bf16* outh = tbh + slot * DD;

    if (ts > bs) {                       // invalid pair -> zeros
        for (int c = 0; c < CHN; ++c) {
            out[c * HWp + idx]  = 0.f;
            outh[c * HWp + idx] = (__bf16)0.f;
        }
        return;
    }

    const float dx = sh[0], dy = sh[1], cs = sh[2], sn = sh[3];
    const float xs = (2.f * (float)x + 1.f) / (float)WWp - 1.f;
    const float ys = (2.f * (float)y + 1.f) / (float)HH  - 1.f;
    const float gx =  cs * xs + sn * ys + dx;
    const float gy = -sn * xs + cs * ys + dy;
    const float ix = ((gx + 1.f) * (float)WWp - 1.f) * 0.5f;
    const float iy = ((gy + 1.f) * (float)HH  - 1.f) * 0.5f;
    const float ix0f = floorf(ix), iy0f = floorf(iy);
    const float wx1 = ix - ix0f, wx0 = 1.f - wx1;
    const float wy1 = iy - iy0f, wy0 = 1.f - wy1;
    const int ix0 = (int)ix0f, iy0 = (int)iy0f;
    const int ix1 = ix0 + 1,   iy1 = iy0 + 1;
    const bool bx0 = (ix0 >= 0) && (ix0 < WWp);
    const bool bx1 = (ix1 >= 0) && (ix1 < WWp);
    const bool by0 = (iy0 >= 0) && (iy0 < HH);
    const bool by1 = (iy1 >= 0) && (iy1 < HH);
    const int cx0 = ix0 < 0 ? 0 : (ix0 > WWp - 1 ? WWp - 1 : ix0);
    const int cx1 = ix1 < 0 ? 0 : (ix1 > WWp - 1 ? WWp - 1 : ix1);
    const int cy0 = iy0 < 0 ? 0 : (iy0 > HH  - 1 ? HH  - 1 : iy0);
    const int cy1 = iy1 < 0 ? 0 : (iy1 > HH  - 1 ? HH  - 1 : iy1);
    const float w00 = wx0 * wy0 * (float)(bx0 && by0);
    const float w10 = wx1 * wy0 * (float)(bx1 && by0);
    const float w01 = wx0 * wy1 * (float)(bx0 && by1);
    const float w11 = wx1 * wy1 * (float)(bx1 && by1);

    const float* src = ff + (size_t)ts * DD;
    const int o00 = cy0 * WWp + cx0, o10 = cy0 * WWp + cx1;
    const int o01 = cy1 * WWp + cx0, o11 = cy1 * WWp + cx1;
#pragma unroll 4
    for (int c = 0; c < CHN; ++c) {
        const float* pc = src + c * HWp;
        const float v = pc[o00] * w00 + pc[o10] * w10 +
                        pc[o01] * w01 + pc[o11] * w11;
        out[c * HWp + idx]  = v;
        outh[c * HWp + idx] = (__bf16)v;
    }
}

// ---------------------------------------------------------------- kernel 2:
// tb[b, 9, :] = ff[b, :]  (identity slot)
__global__ void tf_copy_kernel(const float* __restrict__ ff,
                               float* __restrict__ tb,
                               __bf16* __restrict__ tbh) {
    const size_t i = (size_t)blockIdx.x * blockDim.x + threadIdx.x;
    if (i >= (size_t)TT * DD) return;
    const int    b = (int)(i / DD);
    const size_t d = i - (size_t)b * DD;
    const float  v = ff[i];
    const size_t o = ((size_t)b * TT + (TT - 1)) * DD + d;
    tb[o]  = v;
    tbh[o] = (__bf16)v;
}

// ---------------------------------------------------------------- kernel 3:
// scores[b] = tb[b] @ tb[b]^T via v_wmma_f32_16x16x32_bf16.
// One block per batch b; 8 waves stream disjoint K-chunks, each holding TWO
// independent 16x16 f32 accumulators (2x unroll) so the 8 b128 loads of an
// iteration clause ahead of two back-to-back WMMAs; tiles reduced via LDS.
//
// Padding rows: C[m,n] with m>=10 or n>=10 is never read downstream, so
// lanes with m>=10 simply re-load row (m-10) -- no branch, no zero-fill,
// EXEC stays all-1s (WMMA requirement).
//
// A-layout (16x32 MxK): lanes 0-15 = rows M, K 0-15 contiguous;
//                       lanes 16-31 = rows M, K 16-31 contiguous.
// B-layout (32x16 KxN): lanes 0-15 = cols N, K 0-15 contiguous;
//                       lanes 16-31 = cols N, K 16-31 contiguous.
__global__ void tf_gram_kernel(const __bf16* __restrict__ tbh,
                               float* __restrict__ scores) {
    const int b    = blockIdx.x;
    const int lane = threadIdx.x & 31;
    const int wave = threadIdx.x >> 5;        // 0..7
    const int half = lane >> 4;               // lane group
    const int m    = lane & 15;               // row (A) == col (B)
    const int row  = (m < TT) ? m : (m - TT); // pad lanes alias a valid row

    const __bf16* rowp = tbh + ((size_t)b * TT + row) * DD;
    const __bf16* pa   = rowp + half * 16;    // A image base (2 x 16B per chunk)
    const __bf16* pb   = rowp + half * 16;    // B image base (1 x 32B per chunk)

    v8f c0 = {}, c1 = {};
    // 2112 chunks of K=32; 8 waves x (2 chunks/iter) -> stride 512 elements.
    for (int d0 = wave * 32; d0 < DD; d0 += 8 * 32 * 2) {
        v16bf a0, a1;
        {
            const v8bf lo = *(const v8bf*)(pa + d0);
            const v8bf hi = *(const v8bf*)(pa + d0 + 8);
#pragma unroll
            for (int i = 0; i < 8; ++i) { a0[i] = lo[i]; a0[i + 8] = hi[i]; }
        }
        {
            const v8bf lo = *(const v8bf*)(pa + d0 + 256);
            const v8bf hi = *(const v8bf*)(pa + d0 + 256 + 8);
#pragma unroll
            for (int i = 0; i < 8; ++i) { a1[i] = lo[i]; a1[i + 8] = hi[i]; }
        }
        const v16bf b0 = *(const v16bf*)(pb + d0);
        const v16bf b1 = *(const v16bf*)(pb + d0 + 256);
        c0 = __builtin_amdgcn_wmma_f32_16x16x32_bf16(
                 false, a0, false, b0, (short)0, c0, false, false);
        c1 = __builtin_amdgcn_wmma_f32_16x16x32_bf16(
                 false, a1, false, b1, (short)0, c1, false, false);
    }

    __shared__ float acc[256];
    acc[threadIdx.x] = 0.f;
    __syncthreads();
#pragma unroll
    for (int r = 0; r < 8; ++r) {
        const int crow = r + 8 * half;        // C layout: VGPR r -> M = r + 8*halfSel
        atomicAdd(&acc[crow * 16 + m], c0[r] + c1[r]);
    }
    __syncthreads();
    scores[b * 256 + threadIdx.x] = acc[threadIdx.x];
}

// ---------------------------------------------------------------- kernel 4:
// softmax over s, then collapse mean over t:  w[b,s] = (1/10) sum_t attn[b,t,s]
__global__ void tf_attn_kernel(const float* __restrict__ scores,
                               float* __restrict__ wgt) {
    __shared__ float wsh[TT * TT];
    const int t = threadIdx.x;                 // 128 threads
    if (t < TT * TT) wsh[t] = 0.f;
    __syncthreads();
    if (t < TT * TT) {
        const int b = t / TT, row = t % TT;
        const float* s = scores + b * 256 + row * 16;
        float mx = -1e30f;
        for (int i = 0; i < TT; ++i) mx = fmaxf(mx, s[i]);
        float e[TT];
        float sum = 0.f;
        for (int i = 0; i < TT; ++i) { e[i] = __expf(s[i] - mx); sum += e[i]; }
        const float inv = 1.f / (sum * (float)TT);
        for (int i = 0; i < TT; ++i) atomicAdd(&wsh[b * TT + i], e[i] * inv);
    }
    __syncthreads();
    if (t < TT * TT) wgt[t] = wsh[t];
}

// ---------------------------------------------------------------- kernel 5:
// out[b, d] = sum_s w[b,s] * tb[b, s, d]   (f32 tb for full precision)
__global__ void tf_out_kernel(const float* __restrict__ tb,
                              const float* __restrict__ wgt,
                              float* __restrict__ out) {
    __shared__ float wsh[TT * TT];
    if (threadIdx.x < TT * TT) wsh[threadIdx.x] = wgt[threadIdx.x];
    __syncthreads();
    const size_t i = (size_t)blockIdx.x * blockDim.x + threadIdx.x;
    if (i >= (size_t)TT * DD) return;
    const int    b = (int)(i / DD);
    const size_t d = i - (size_t)b * DD;
    const float* base = tb + (size_t)b * TT * DD + d;
    float acc = 0.f;
#pragma unroll
    for (int s = 0; s < TT; ++s) acc += wsh[b * TT + s] * base[(size_t)s * DD];
    out[i] = acc;
}

// ---------------------------------------------------------------- launcher
extern "C" void kernel_launch(void* const* d_in, const int* in_sizes, int n_in,
                              void* d_out, int out_size, void* d_ws, size_t ws_size,
                              hipStream_t stream) {
    const float* ff = (const float*)d_in[0];   // (10, 256, 264) f32
    const float* dp = (const float*)d_in[1];   // (10, 3) f32
    float* out = (float*)d_out;                // (10, 256, 264) f32

    char* ws = (char*)d_ws;
    float*  tb     = (float*)(ws);
    __bf16* tbh    = (__bf16*)(ws + TB_BF16_OFF);
    float*  scores = (float*)(ws + SCORES_OFF);
    float*  wgt    = (float*)(ws + WGT_OFF);

    // 90 warp pairs x 4224 pixels
    tf_warp_kernel<<<dim3((HWp + 255) / 256, TT * (TT - 1)), 256, 0, stream>>>(
        ff, dp, tb, tbh);
    // identity slot
    tf_copy_kernel<<<(TT * DD + 255) / 256, 256, 0, stream>>>(ff, tb, tbh);
    // WMMA gram per batch
    tf_gram_kernel<<<TT, 256, 0, stream>>>(tbh, scores);
    // softmax + mean-collapse to per-row weights
    tf_attn_kernel<<<1, 128, 0, stream>>>(scores, wgt);
    // weighted row-sum output
    tf_out_kernel<<<(TT * DD + 255) / 256, 256, 0, stream>>>(tb, wgt, out);
}
